// MultiHeadCrossAttention_66683662238494
// MI455X (gfx1250) — matrix operand order repaired and checked by reading, not compiled, not run on hardware
//
#include <hip/hip_runtime.h>
#include <hip/hip_fp16.h>

// ---------------------------------------------------------------------------
// MultiHeadCrossAttention for MI455X (gfx1250).
// All GEMMs via v_wmma_f32_16x16x32_f16; tile staging via Tensor Data Mover
// (tensor_load_to_lds) DOUBLE-BUFFERED so DMA overlaps the WMMA bursts;
// completion via s_wait_tensorcnt behind workgroup barriers.
// B=4, Lq=2048, Lk=1024, DIM=1024, H=16, HD=64.
// ---------------------------------------------------------------------------

#define DIMC   1024
#define NHEADS 16
#define HD     64
#define GAMMA_ 10.0f

typedef __attribute__((ext_vector_type(16))) _Float16 v16h;
typedef __attribute__((ext_vector_type(8)))  _Float16 v8h;
typedef __attribute__((ext_vector_type(8)))  float    v8f;
typedef __attribute__((ext_vector_type(4)))  float    v4f;
typedef __attribute__((ext_vector_type(4)))  unsigned int u32x4;
typedef __attribute__((ext_vector_type(8)))  int      i32x8;
typedef __attribute__((ext_vector_type(4)))  int      i32x4;

#if __has_builtin(__builtin_amdgcn_tensor_load_to_lds)
#define HAVE_TDM 1
#else
#define HAVE_TDM 0
#endif

// ---------------- helpers ---------------------------------------------------

__device__ __forceinline__ v8f wmma_f16(v16h a, v16h b, v8f c) {
  // D = A(16x32 f16) * B(32x16 f16) + C(16x16 f32)
  return __builtin_amdgcn_wmma_f32_16x16x32_f16(
      /*neg_a=*/false, a, /*neg_b=*/false, b,
      /*c_mod=*/(short)0, c, /*reuse_a=*/false, /*reuse_b=*/false);
}

__device__ __forceinline__ v16h cat8(v8h lo, v8h hi) {
  v16h r;
#pragma unroll
  for (int i = 0; i < 8; ++i) { r[i] = lo[i]; r[i + 8] = hi[i]; }
  return r;
}

// convert 8 contiguous fp32 (32B-aligned) to 8 halves
__device__ __forceinline__ v8h cvt8(const float* __restrict__ p) {
  v4f a = *(const v4f*)p;
  v4f b = *(const v4f*)(p + 4);
  v8h r;
#pragma unroll
  for (int i = 0; i < 4; ++i) { r[i] = (_Float16)a[i]; r[i + 4] = (_Float16)b[i]; }
  return r;
}

__device__ __forceinline__ float redmax16(float v) {
  v = fmaxf(v, __shfl_xor(v, 1));
  v = fmaxf(v, __shfl_xor(v, 2));
  v = fmaxf(v, __shfl_xor(v, 4));
  v = fmaxf(v, __shfl_xor(v, 8));
  return v;
}
__device__ __forceinline__ float redsum16(float v) {
  v += __shfl_xor(v, 1);
  v += __shfl_xor(v, 2);
  v += __shfl_xor(v, 4);
  v += __shfl_xor(v, 8);
  return v;
}

#if HAVE_TDM
// Issue a 2-D TDM tile load (f16 elements) into LDS. One call per wave.
// tile: tile_h rows x tile_w elems, row stride = stride0 elems, row-major into LDS.
// clang-23 signature: (u32x4 g0, i32x8 g1, i32x4 g2, i32x4 g3, i32x8 pad, i32 cpol)
__device__ __forceinline__ void tdm_load_2d_f16(unsigned lds_byte_addr,
                                                const _Float16* gptr,
                                                unsigned tile_w, unsigned tile_h,
                                                unsigned stride0) {
  unsigned long long ga = (unsigned long long)(uintptr_t)gptr;
  u32x4 g0;
  g0[0] = 1u;                                         // count = 1 valid descriptor
  g0[1] = lds_byte_addr;                              // lds_addr
  g0[2] = (unsigned)(ga & 0xFFFFFFFFu);               // global_addr[31:0]
  g0[3] = (unsigned)((ga >> 32) & 0x01FFFFFFu) | (2u << 30);  // addr[56:32] | type=2
  i32x8 g1;
  g1[0] = (int)(1u << 16);                            // data_size=1 (2 bytes/elem)
  g1[1] = (int)((tile_w & 0xFFFFu) << 16);            // tensor_dim0[15:0]
  g1[2] = (int)((tile_w >> 16) | ((tile_h & 0xFFFFu) << 16));  // dim0 hi | dim1 lo
  g1[3] = (int)((tile_h >> 16) | (tile_w << 16));     // dim1 hi | tile_dim0
  g1[4] = (int)tile_h;                                // tile_dim1 (tile_dim2 = 0)
  g1[5] = (int)stride0;                               // tensor_dim0_stride[31:0]
  g1[6] = 0;
  g1[7] = 0;
  i32x4 z4 = {0, 0, 0, 0};
  i32x8 z8 = {0, 0, 0, 0, 0, 0, 0, 0};
  __builtin_amdgcn_tensor_load_to_lds(g0, g1, z4, z4, z8, 0);
}
#endif

__device__ __forceinline__ unsigned lds_addr32(const void* p) {
  return (unsigned)(uintptr_t)p;   // low 32 bits map to LDS offset (aperture rule)
}

// ---------------- kernel 1: W fp32 [n,k] -> f16 transposed [k,n] -------------

__global__ void __launch_bounds__(256)
wt_convert_kernel(const float* __restrict__ W, _Float16* __restrict__ Wt) {
  int i = blockIdx.x * 256 + threadIdx.x;        // over 1024*1024
  int n = i >> 10, k = i & 1023;
  Wt[(size_t)k * DIMC + n] = (_Float16)W[i];
}

// ---------------- kernel 2: projection + bias + RoPE -------------------------
// Y = X @ W^T + b ; block = 256 threads = 8 waves; wave tile = 16 rows x 1 head.
// Double-buffered TDM staging of 64x64 W^T tiles (k-step 64 -> 8 WMMA / stage).
// mode: 0 = V (no rope, [B,H,L,HD]); 1 = Q (rope, [B,H,L,HD]); 2 = K (rope, [B,H,HD,L])

__global__ void __launch_bounds__(256)
proj_rope_kernel(const float* __restrict__ X, const _Float16* __restrict__ Wt,
                 const float* __restrict__ bias, _Float16* __restrict__ Y,
                 int L, int mode) {
  __shared__ __align__(32) _Float16 lds_w[2][64 * 64];  // ping-pong 64k x 64n tiles

  const int tid = threadIdx.x, lane = tid & 31, wave = tid >> 5;
  const int rowbase = blockIdx.x * 128 + wave * 16;  // flattened row (b*L + t)
  const int head = blockIdx.y;
  const int l15 = lane & 15, lhi = lane >> 4;        // lhi in {0,1}
  const float* xrow = X + (size_t)(rowbase + l15) * DIMC;
  const int ko8 = lhi * 8;

  v8f acc[4];
#pragma unroll
  for (int c = 0; c < 4; ++c)
#pragma unroll
    for (int j = 0; j < 8; ++j) acc[c][j] = 0.0f;

#if HAVE_TDM
  // prologue: stage tile 0
  if (wave == 0) {
    tdm_load_2d_f16(lds_addr32(lds_w[0]), Wt + head * HD, 64, 64, DIMC);
    __builtin_amdgcn_s_wait_tensorcnt(0);
  }
  __syncthreads();
#endif

  for (int k0 = 0; k0 < DIMC; k0 += 64) {
    const int cur = (k0 >> 6) & 1;
#if HAVE_TDM
    // kick off next tile's DMA before computing on this one (overlap)
    if (wave == 0 && k0 + 64 < DIMC) {
      tdm_load_2d_f16(lds_addr32(lds_w[cur ^ 1]),
                      Wt + (size_t)(k0 + 64) * DIMC + head * HD, 64, 64, DIMC);
    }
#else
    __syncthreads();
    for (int i = tid; i < 64 * 8; i += 256) {
      int rr = i >> 3, cc = i & 7;
      *(v8h*)&lds_w[cur][rr * 64 + cc * 8] =
          *(const v8h*)(Wt + (size_t)(k0 + rr) * DIMC + head * HD + cc * 8);
    }
    __syncthreads();
#endif
    const _Float16* wbuf = lds_w[cur];

    // A fragments (16x32 f16) converted on the fly from fp32 X.
    // lane<16: K = {0..7, 16..23}; lane>=16: K = {8..15, 24..31}
    v16h a0 = cat8(cvt8(xrow + k0 + ko8),      cvt8(xrow + k0 + 16 + ko8));
    v16h a1 = cat8(cvt8(xrow + k0 + 32 + ko8), cvt8(xrow + k0 + 48 + ko8));
    if (k0 + 64 < DIMC) __builtin_prefetch(xrow + k0 + 64, 0, 1);

#pragma unroll
    for (int c = 0; c < 4; ++c) {
      // B fragments (32x16): lane%16 = N column, K in table order per lane
      v16h b0, b1;
#pragma unroll
      for (int s = 0; s < 16; ++s) {
        int kk = (s < 8) ? (ko8 + s) : (8 + ko8 + s);
        b0[s] = wbuf[kk * 64 + c * 16 + l15];
        b1[s] = wbuf[(32 + kk) * 64 + c * 16 + l15];
      }
      acc[c] = wmma_f16(a0, b0, acc[c]);
      acc[c] = wmma_f16(a1, b1, acc[c]);
    }

#if HAVE_TDM
    __syncthreads();                        // everyone done reading buffer `cur`
    if (wave == 0) __builtin_amdgcn_s_wait_tensorcnt(0);  // next buffer landed
    __syncthreads();
#endif
  }

  // bias
#pragma unroll
  for (int c = 0; c < 4; ++c) {
    float bv = bias[head * HD + c * 16 + l15];
#pragma unroll
    for (int j = 0; j < 8; ++j) acc[c][j] += bv;
  }

  const int b = rowbase / L;
  const int tbase = rowbase % L;

  // RoPE: pair (d, d+32) -> accumulator chunks (c, c+2)
  if (mode != 0) {
#pragma unroll
    for (int c = 0; c < 2; ++c) {
      int d1 = c * 16 + l15;  // 0..31
      float fr = __expf((float)d1 * (-9.2103403720f / 32.0f));  // 10000^(-d/32)
#pragma unroll
      for (int j = 0; j < 8; ++j) {
        int t = tbase + j + lhi * 8;
        float th = GAMMA_ * ((float)t / (float)L) * fr;
        float sn, cs;
        __sincosf(th, &sn, &cs);
        float x1 = acc[c][j], x2 = acc[c + 2][j];
        acc[c][j]     = x1 * cs - x2 * sn;
        acc[c + 2][j] = x1 * sn + x2 * cs;
      }
    }
  }

  if (mode == 2) {
    // K stored transposed: [B,H,HD,L]
#pragma unroll
    for (int c = 0; c < 4; ++c) {
      int d = c * 16 + l15;
      size_t base = (((size_t)b * NHEADS + head) * HD + d) * (size_t)L + tbase + lhi * 8;
      v8h pk;
#pragma unroll
      for (int j = 0; j < 8; ++j) pk[j] = (_Float16)acc[c][j];
      *(v8h*)&Y[base] = pk;
    }
  } else {
    // Q/V stored [B,H,L,HD]
#pragma unroll
    for (int c = 0; c < 4; ++c) {
      int d = c * 16 + l15;
#pragma unroll
      for (int j = 0; j < 8; ++j) {
        int t = tbase + j + lhi * 8;
        Y[(((size_t)b * NHEADS + head) * L + t) * HD + d] = (_Float16)acc[c][j];
      }
    }
  }
}

// ---------------- kernel 3: flash attention ---------------------------------
// block = 128 threads (4 waves); block tile = (b, h, 64 q rows); wave = 16 q rows.
// Double-buffered TDM staging of K^T and V 64x64 tiles.

__global__ void __launch_bounds__(128)
flash_attn_kernel(const _Float16* __restrict__ q16,
                  const _Float16* __restrict__ kT16,  // [B,H,HD,Lk]
                  const _Float16* __restrict__ v16,   // [B,H,Lk,HD]
                  _Float16* __restrict__ attn16,      // [B,H,Lq,HD]
                  int Lq, int Lk) {
  __shared__ __align__(32) _Float16 lds_k[2][64 * 64];  // [d][t] tiles of K^T
  __shared__ __align__(32) _Float16 lds_v[2][64 * 64];  // [t][d] tiles of V
  __shared__ __align__(32) _Float16 p_lds[4][16][64];   // per-wave P relayout buffer

  const int tid = threadIdx.x, lane = tid & 31, wave = tid >> 5;
  const int l15 = lane & 15, lhi = lane >> 4;
  const int bh = blockIdx.z * NHEADS + blockIdx.y;
  const int qbase = blockIdx.x * 64 + wave * 16;
  const float scale = 0.125f;  // 1/sqrt(64)

  // Q A-fragments for this wave's 16 rows: K-dim = d (0..63) -> two 16x32 frags
  const _Float16* qp = q16 + ((size_t)bh * Lq + qbase + l15) * HD;
  const int ko8 = lhi * 8;
  v16h aq0 = cat8(*(const v8h*)(qp + ko8),      *(const v8h*)(qp + 16 + ko8));
  v16h aq1 = cat8(*(const v8h*)(qp + 32 + ko8), *(const v8h*)(qp + 48 + ko8));

  const _Float16* kTp = kT16 + (size_t)bh * HD * Lk;
  const _Float16* vp  = v16  + (size_t)bh * Lk * HD;

  float rm[8], rl[8];
  v8f o[4];
#pragma unroll
  for (int j = 0; j < 8; ++j) { rm[j] = -1e30f; rl[j] = 0.0f; }
#pragma unroll
  for (int c = 0; c < 4; ++c)
#pragma unroll
    for (int j = 0; j < 8; ++j) o[c][j] = 0.0f;

#if HAVE_TDM
  // prologue: stage K/V tile 0
  if (wave == 0) {
    tdm_load_2d_f16(lds_addr32(lds_k[0]), kTp, 64, 64, (unsigned)Lk);
    tdm_load_2d_f16(lds_addr32(lds_v[0]), vp, 64, 64, HD);
    __builtin_amdgcn_s_wait_tensorcnt(0);
  }
  __syncthreads();
#endif

  for (int kt0 = 0; kt0 < Lk; kt0 += 64) {
    const int cur = (kt0 >> 6) & 1;
#if HAVE_TDM
    // issue next K/V tiles while we compute on the current ones
    if (wave == 0 && kt0 + 64 < Lk) {
      tdm_load_2d_f16(lds_addr32(lds_k[cur ^ 1]), kTp + kt0 + 64, 64, 64, (unsigned)Lk);
      tdm_load_2d_f16(lds_addr32(lds_v[cur ^ 1]), vp + (size_t)(kt0 + 64) * HD, 64, 64, HD);
    }
#else
    __syncthreads();
    for (int i = tid; i < 64 * 8; i += 128) {
      int rr = i >> 3, cc = i & 7;
      *(v8h*)&lds_k[cur][rr * 64 + cc * 8] =
          *(const v8h*)(kTp + (size_t)rr * Lk + kt0 + cc * 8);
      *(v8h*)&lds_v[cur][rr * 64 + cc * 8] =
          *(const v8h*)(vp + (size_t)(kt0 + rr) * HD + cc * 8);
    }
    __syncthreads();
#endif
    const _Float16* kbuf = lds_k[cur];
    const _Float16* vbuf = lds_v[cur];

    // ---- S = scale * Q K^T : 16 q rows x 64 k cols -------------------------
    v8f s[4];
#pragma unroll
    for (int c = 0; c < 4; ++c) {
      v8f z;
#pragma unroll
      for (int j = 0; j < 8; ++j) z[j] = 0.0f;
      v16h bk0, bk1;  // K=d 0..31 / 32..63
#pragma unroll
      for (int s = 0; s < 16; ++s) {
        int kk = (s < 8) ? (ko8 + s) : (8 + ko8 + s);
        bk0[s] = kbuf[kk * 64 + c * 16 + l15];
        bk1[s] = kbuf[(32 + kk) * 64 + c * 16 + l15];
      }
      z = wmma_f16(aq0, bk0, z);
      z = wmma_f16(aq1, bk1, z);
#pragma unroll
      for (int j = 0; j < 8; ++j) z[j] *= scale;
      s[c] = z;
    }

    // ---- online softmax update --------------------------------------------
#pragma unroll
    for (int j = 0; j < 8; ++j) {
      float mx = rm[j];
#pragma unroll
      for (int c = 0; c < 4; ++c) mx = fmaxf(mx, s[c][j]);
      mx = redmax16(mx);
      float corr = __expf(rm[j] - mx);
      rm[j] = mx;
      float rs = 0.0f;
#pragma unroll
      for (int c = 0; c < 4; ++c) {
        float p = __expf(s[c][j] - mx);
        s[c][j] = p;
        rs += p;
      }
      rs = redsum16(rs);
      rl[j] = rl[j] * corr + rs;
#pragma unroll
      for (int c = 0; c < 4; ++c) o[c][j] *= corr;
    }

    // ---- relayout P (C-layout) -> A-fragments through per-wave LDS ---------
#pragma unroll
    for (int c = 0; c < 4; ++c)
#pragma unroll
      for (int j = 0; j < 8; ++j)
        p_lds[wave][j + lhi * 8][c * 16 + l15] = (_Float16)s[c][j];

    v16h p0 = cat8(*(const v8h*)&p_lds[wave][l15][ko8],
                   *(const v8h*)&p_lds[wave][l15][16 + ko8]);
    v16h p1 = cat8(*(const v8h*)&p_lds[wave][l15][32 + ko8],
                   *(const v8h*)&p_lds[wave][l15][48 + ko8]);

    // ---- O += P V ----------------------------------------------------------
#pragma unroll
    for (int c = 0; c < 4; ++c) {
      v16h bv0, bv1;  // K=t 0..31 / 32..63
#pragma unroll
      for (int s = 0; s < 16; ++s) {
        int kk = (s < 8) ? (ko8 + s) : (8 + ko8 + s);
        bv0[s] = vbuf[kk * 64 + c * 16 + l15];
        bv1[s] = vbuf[(32 + kk) * 64 + c * 16 + l15];
      }
      o[c] = wmma_f16(p0, bv0, o[c]);
      o[c] = wmma_f16(p1, bv1, o[c]);
    }

#if HAVE_TDM
    __syncthreads();                        // everyone done with buffer `cur`
    if (wave == 0) __builtin_amdgcn_s_wait_tensorcnt(0);  // next tiles landed
    __syncthreads();
#endif
  }

  // ---- normalize + store f16 [B,H,Lq,HD] -----------------------------------
#pragma unroll
  for (int c = 0; c < 4; ++c) {
    int d = c * 16 + l15;
#pragma unroll
    for (int j = 0; j < 8; ++j) {
      int t = qbase + j + lhi * 8;
      attn16[((size_t)bh * Lq + t) * HD + d] = (_Float16)(o[c][j] / rl[j]);
    }
  }
}

// ---------------- kernel 4: output projection (f32 out) ----------------------
// Out[B*Lq,1024] = Attn @ Wo^T + bo ; same double-buffered tiling as proj.

__global__ void __launch_bounds__(256)
oproj_kernel(const _Float16* __restrict__ A16,   // [B,H,Lq,HD]
             const _Float16* __restrict__ WoT,   // [k,n] f16
             const float* __restrict__ bias, float* __restrict__ Out, int Lq) {
  __shared__ __align__(32) _Float16 lds_w[2][64 * 64];

  const int tid = threadIdx.x, lane = tid & 31, wave = tid >> 5;
  const int rowbase = blockIdx.x * 128 + wave * 16;  // row in [0, B*Lq)
  const int ngrp = blockIdx.y;                       // 64-wide output column group
  const int l15 = lane & 15, lhi = lane >> 4;
  const int b = rowbase / Lq, tbase = rowbase % Lq;
  const int ko8 = lhi * 8;

  v8f acc[4];
#pragma unroll
  for (int c = 0; c < 4; ++c)
#pragma unroll
    for (int j = 0; j < 8; ++j) acc[c][j] = 0.0f;

#if HAVE_TDM
  if (wave == 0) {
    tdm_load_2d_f16(lds_addr32(lds_w[0]), WoT + ngrp * 64, 64, 64, DIMC);
    __builtin_amdgcn_s_wait_tensorcnt(0);
  }
  __syncthreads();
#endif

  for (int k0 = 0; k0 < DIMC; k0 += 64) {
    const int cur = (k0 >> 6) & 1;
#if HAVE_TDM
    if (wave == 0 && k0 + 64 < DIMC) {
      tdm_load_2d_f16(lds_addr32(lds_w[cur ^ 1]),
                      WoT + (size_t)(k0 + 64) * DIMC + ngrp * 64, 64, 64, DIMC);
    }
#else
    __syncthreads();
    for (int i = tid; i < 64 * 8; i += 256) {
      int rr = i >> 3, cc = i & 7;
      *(v8h*)&lds_w[cur][rr * 64 + cc * 8] =
          *(const v8h*)(WoT + (size_t)(k0 + rr) * DIMC + ngrp * 64 + cc * 8);
    }
    __syncthreads();
#endif
    const _Float16* wbuf = lds_w[cur];

    // A-fragments: k index = h*64 + dd maps into [B,H,Lq,HD] storage
    // (k0 step 64 == exactly one head h)
    int h = k0 >> 6;
    const _Float16* ap =
        A16 + (((size_t)b * NHEADS + h) * Lq + tbase + l15) * HD;
    v16h a0 = cat8(*(const v8h*)(ap + ko8),      *(const v8h*)(ap + 16 + ko8));
    v16h a1 = cat8(*(const v8h*)(ap + 32 + ko8), *(const v8h*)(ap + 48 + ko8));

#pragma unroll
    for (int c = 0; c < 4; ++c) {
      v16h b0, b1;
#pragma unroll
      for (int s = 0; s < 16; ++s) {
        int kk = (s < 8) ? (ko8 + s) : (8 + ko8 + s);
        b0[s] = wbuf[kk * 64 + c * 16 + l15];
        b1[s] = wbuf[(32 + kk) * 64 + c * 16 + l15];
      }
      acc[c] = wmma_f16(a0, b0, acc[c]);
      acc[c] = wmma_f16(a1, b1, acc[c]);
    }

#if HAVE_TDM
    __syncthreads();
    if (wave == 0) __builtin_amdgcn_s_wait_tensorcnt(0);
    __syncthreads();
#endif
  }

#pragma unroll
  for (int c = 0; c < 4; ++c) {
    int n = ngrp * 64 + c * 16 + l15;
    float bv = bias[n];
#pragma unroll
    for (int j = 0; j < 8; ++j) {
      int row = rowbase + j + lhi * 8;
      Out[(size_t)row * DIMC + n] = acc[c][j] + bv;
    }
  }
}

// ---------------- launcher ---------------------------------------------------

extern "C" void kernel_launch(void* const* d_in, const int* in_sizes, int n_in,
                              void* d_out, int out_size, void* d_ws, size_t ws_size,
                              hipStream_t stream) {
  (void)in_sizes; (void)n_in; (void)out_size; (void)ws_size;

  const int B = 4, Lq = 2048, Lk = 1024;

  const float* query = (const float*)d_in[0];
  const float* key   = (const float*)d_in[1];
  const float* value = (const float*)d_in[2];
  const float* Wq = (const float*)d_in[3];
  const float* bq = (const float*)d_in[4];
  const float* Wk = (const float*)d_in[5];
  const float* bk = (const float*)d_in[6];
  const float* Wv = (const float*)d_in[7];
  const float* bv = (const float*)d_in[8];
  const float* Wo = (const float*)d_in[9];
  const float* bo = (const float*)d_in[10];

  // workspace layout (in f16 elements); total 28 * 2^20 halves = 56 MiB
  const size_t MM = (size_t)1024 * 1024;
  _Float16* w = (_Float16*)d_ws;
  _Float16* WqT    = w;             // 1 MM
  _Float16* WkT    = w + 1 * MM;    // 1 MM
  _Float16* WvT    = w + 2 * MM;    // 1 MM
  _Float16* WoT    = w + 3 * MM;    // 1 MM
  _Float16* q16    = w + 4 * MM;    // B*Lq*DIM = 8 MM
  _Float16* kT16   = w + 12 * MM;   // B*DIM*Lk = 4 MM
  _Float16* v16    = w + 16 * MM;   // 4 MM
  _Float16* attn16 = w + 20 * MM;   // 8 MM

  // 1) weight transpose + f16 convert
  wt_convert_kernel<<<4096, 256, 0, stream>>>(Wq, WqT);
  wt_convert_kernel<<<4096, 256, 0, stream>>>(Wk, WkT);
  wt_convert_kernel<<<4096, 256, 0, stream>>>(Wv, WvT);
  wt_convert_kernel<<<4096, 256, 0, stream>>>(Wo, WoT);

  // 2) projections (+bias, +RoPE for Q/K; K stored transposed)
  proj_rope_kernel<<<dim3((B * Lq) / 128, NHEADS), 256, 0, stream>>>(
      query, WqT, bq, q16, Lq, /*mode=*/1);
  proj_rope_kernel<<<dim3((B * Lk) / 128, NHEADS), 256, 0, stream>>>(
      key, WkT, bk, kT16, Lk, /*mode=*/2);
  proj_rope_kernel<<<dim3((B * Lk) / 128, NHEADS), 256, 0, stream>>>(
      value, WvT, bv, v16, Lk, /*mode=*/0);

  // 3) flash attention
  flash_attn_kernel<<<dim3(Lq / 64, NHEADS, B), 128, 0, stream>>>(
      q16, kT16, v16, attn16, Lq, Lk);

  // 4) output projection -> fp32 d_out
  oproj_kernel<<<dim3((B * Lq) / 128, NHEADS), 256, 0, stream>>>(
      attn16, WoT, bo, (float*)d_out, Lq);
}
